// CODANO_35244501631604
// MI455X (gfx1250) — matrix operand order repaired and checked by reading, not compiled, not run on hardware
//
#include <hip/hip_runtime.h>
#include <hip/hip_bf16.h>

// ---------------------------------------------------------------------------
// CoDA-NO forward for MI455X (gfx1250, wave32).
// Spectral convs = truncated-DFT matmuls on V_WMMA_F32_16X16X32_F16.
// Twiddle tiles staged to LDS via the Tensor Data Mover (padded rows).
// ---------------------------------------------------------------------------

typedef __attribute__((ext_vector_type(16))) _Float16 v16h;
typedef __attribute__((ext_vector_type(4)))  _Float16 v4h;
typedef __attribute__((ext_vector_type(8)))  float    v8f;
typedef __attribute__((ext_vector_type(4)))  float    f32x4;
typedef __attribute__((ext_vector_type(4)))  unsigned int u32x4;
typedef __attribute__((ext_vector_type(8)))  int      i32x8;
typedef __attribute__((ext_vector_type(4)))  int      i32x4;

#define HW 16384
#define NB 8
#define NV 5
#define NT 40     // tokens
#define CC 4      // token dim
#define HC 8      // heads*token dim

__device__ __forceinline__ v8f wmma_f16(v16h a, v16h b, v8f c) {
  return __builtin_amdgcn_wmma_f32_16x16x32_f16(false, a, false, b, (short)0, c,
                                                false, false);
}

// K index for element e of the 16-bit A fragment (16x32), contiguous k = 16*half + e (matches B).
__device__ __forceinline__ int akidx(int e, int half) {
  return half * 16 + e;
}

__device__ __forceinline__ v16h ldsA(const _Float16* s, int ld) {
  int lane = threadIdx.x & 31;
  int m = lane & 15, half = lane >> 4;
  v16h a;
#pragma unroll
  for (int e = 0; e < 16; ++e) a[e] = s[m * ld + akidx(e, half)];
  return a;
}

// B fragment (32x16), source laid out [k][n].
__device__ __forceinline__ v16h ldsB(const _Float16* s, int ld, int n0, bool neg) {
  int lane = threadIdx.x & 31;
  int n = n0 + (lane & 15);
  int kb = (lane >> 4) * 16;
  v16h b;
#pragma unroll
  for (int e = 0; e < 16; ++e) {
    _Float16 v = s[(kb + e) * ld + n];
    b[e] = neg ? (_Float16)(-v) : v;
  }
  return b;
}

// B fragment from token-major layout [n][k] (i.e. implicit transpose).
__device__ __forceinline__ v16h ldsBt(const _Float16* s, int ld, int n0) {
  int lane = threadIdx.x & 31;
  int n = n0 + (lane & 15);
  int kb = (lane >> 4) * 16;
  v16h b;
#pragma unroll
  for (int e = 0; e < 16; ++e) b[e] = s[n * ld + kb + e];
  return b;
}

__device__ __forceinline__ void stC(float* C, int ldc, int m0, int n0, v8f acc) {
  int lane = threadIdx.x & 31;
  int n = n0 + (lane & 15);
  int mh = (lane >> 4) * 8;
#pragma unroll
  for (int r = 0; r < 8; ++r) C[(size_t)(m0 + mh + r) * ldc + n] = acc[r];
}

__device__ __forceinline__ float gelu_f(float x) {
  float x3 = x * x * x;
  return 0.5f * x * (1.f + tanhf(0.7978845608f * (x + 0.044715f * x3)));
}

// ------------------------- Tensor Data Mover helper -------------------------
// 2D f16 tile load: global (row-major, stride_elems) -> LDS, optional row pad.
// pad_interval_code: pad after 2^(c+1) dwords; pad_amount_code: (c+1) dwords.
__device__ __forceinline__ void tdm_load_2d_f16(const _Float16* gsrc, void* lds_dst,
                                                int tile_w, int tile_h,
                                                int stride_elems, int tensor_h,
                                                int pad_interval_code,
                                                int pad_amount_code, bool pad_en) {
  unsigned long long ga = (unsigned long long)(size_t)gsrc;
  unsigned la = (unsigned)(unsigned long long)(size_t)lds_dst;  // LDS offset = addr[31:0]
  u32x4 g0;
  g0[0] = 1u;                                   // count=1, user descriptor
  g0[1] = la;                                   // lds_addr
  g0[2] = (unsigned)ga;                         // global_addr[31:0]
  g0[3] = (unsigned)((ga >> 32) & 0x01FFFFFFull) | (2u << 30);  // addr[56:32] | type=2
  i32x8 g1;
  unsigned d0 = (1u << 16);                     // data_size = 1 (2 bytes)
  if (pad_en)
    d0 |= (1u << 20) | ((unsigned)pad_interval_code << 22)
        | ((unsigned)pad_amount_code << 25);
  g1[0] = (int)d0;
  g1[1] = (int)(((unsigned)stride_elems & 0xFFFFu) << 16);          // tensor_dim0 lo16
  g1[2] = (int)((((unsigned)stride_elems >> 16) & 0xFFFFu)
              | (((unsigned)tensor_h & 0xFFFFu) << 16));            // dim0 hi | dim1 lo
  g1[3] = (int)((((unsigned)tensor_h >> 16) & 0xFFFFu)
              | (((unsigned)tile_w & 0xFFFFu) << 16));              // dim1 hi | tile_dim0
  g1[4] = (int)((unsigned)tile_h & 0xFFFFu);                        // tile_dim1 (dim2=0)
  g1[5] = (int)(unsigned)stride_elems;                              // tensor_dim0_stride
  g1[6] = 0;
  g1[7] = 0;
  i32x4 z4 = {};
  i32x8 z8 = {};
  __builtin_amdgcn_tensor_load_to_lds(g0, g1, z4, z4, z8, 0);
}

// --------------------------- twiddle tables (f16) ---------------------------
// [0)     Tf1   [128][64] : cos(2pi y k/128) | -sin, k<32         (fwd stage1)
// [8192)  Tf2re [64][128] : cos(2pi kx x/128), kx in {0..31,96..127}
// [16384) Tf2im [64][128] : -sin
// [24576) Ti1re [128][64] : cos(2pi x kx/128)
// [32768) Ti1im [128][64] : +sin
// [40960) Ti2   [64][128] : wk*cos/HW | -wk*sin/HW               (inv stage2)
__global__ void k_fill_tables(_Float16* tab) {
  int i = blockIdx.x * 256 + threadIdx.x;
  if (i >= 49152) return;
  const float TP = 6.28318530717958647692f;
  float val = 0.f;
  if (i < 8192) {
    int y = i >> 6, k = i & 63;
    val = (k < 32) ? __cosf(TP * ((y * k) & 127) / 128.f)
                   : -__sinf(TP * ((y * (k - 32)) & 127) / 128.f);
  } else if (i < 16384) {
    int j = i - 8192; int r = j >> 7, xx = j & 127;
    int kx = (r < 32) ? r : r + 64;
    val = __cosf(TP * ((kx * xx) & 127) / 128.f);
  } else if (i < 24576) {
    int j = i - 16384; int r = j >> 7, xx = j & 127;
    int kx = (r < 32) ? r : r + 64;
    val = -__sinf(TP * ((kx * xx) & 127) / 128.f);
  } else if (i < 32768) {
    int j = i - 24576; int xx = j >> 6, r = j & 63;
    int kx = (r < 32) ? r : r + 64;
    val = __cosf(TP * ((kx * xx) & 127) / 128.f);
  } else if (i < 40960) {
    int j = i - 32768; int xx = j >> 6, r = j & 63;
    int kx = (r < 32) ? r : r + 64;
    val = __sinf(TP * ((kx * xx) & 127) / 128.f);
  } else {
    int j = i - 40960; int k = j >> 7, y = j & 127;
    if (k < 32) {
      float w = (k == 0) ? 1.f : 2.f;
      val = w * __cosf(TP * ((k * y) & 127) / 128.f) * (1.f / 16384.f);
    } else {
      int kk = k - 32;
      float w = (kk == 0) ? 1.f : 2.f;
      val = -w * __sinf(TP * ((kk * y) & 127) / 128.f) * (1.f / 16384.f);
    }
  }
  tab[i] = (_Float16)val;
}

// ------------------------- batched real WMMA GEMM ---------------------------
// C[ch][M][Nn] = A[ch][M][K](f32) * B[K][Nn](f16).  grid(ch, M/16, Nn/64), 128 thr.
__global__ void k_gemm_rr(const float* __restrict__ A, const _Float16* __restrict__ B,
                          float* __restrict__ C, int M, int K, int Nn) {
  int ch = blockIdx.x, mt = blockIdx.y, nz = blockIdx.z;
  const float* Ac = A + (size_t)ch * M * K;
  float* Cc = C + (size_t)ch * M * Nn;
  __shared__ _Float16 sA[16 * 36];
  __shared__ _Float16 sB[32 * 66];   // TDM: 64 f16 row + 1 dword pad -> stride 66
  int tid = threadIdx.x, wave = tid >> 5;
  int m0 = mt * 16, nb0 = nz * 64;
  v8f acc = {};
  for (int k0 = 0; k0 < K; k0 += 32) {
    __syncthreads();
    {   // stage A: 16x32 f32 -> f16, one f32x4 per thread (128 slots)
      int r = tid >> 3, c4 = (tid & 7) * 4;
      f32x4 v = *(const f32x4*)(Ac + (size_t)(m0 + r) * K + k0 + c4);
      v4h h; h[0] = (_Float16)v[0]; h[1] = (_Float16)v[1];
             h[2] = (_Float16)v[2]; h[3] = (_Float16)v[3];
      *(v4h*)(sA + r * 36 + c4) = h;
    }
    if (wave == 0) {   // stage B tile via TDM: pad 1 dword per 32-dword row
      tdm_load_2d_f16(B + (size_t)k0 * Nn + nb0, sB, 64, 32, Nn, K, 4, 0, true);
      __builtin_amdgcn_s_wait_tensorcnt(0);
    }
    __syncthreads();
    v16h a = ldsA(sA, 36);
    v16h b = ldsB(sB, 66, wave * 16, false);
    acc = wmma_f16(a, b, acc);
  }
  stC(Cc, Nn, m0, nb0 + wave * 16, acc);
}

// ---------------------- batched complex WMMA GEMM ---------------------------
// Z[ch][M][64] = T(M x K complex, f16 planes) * Y[ch][K][64] (re cols 0..31 | im).
// grid(ch, M/16), 128 thr: wave -> (re/im part, 16-col tile).
__global__ void k_cgemm(const _Float16* __restrict__ Tre, const _Float16* __restrict__ Tim,
                        const float* __restrict__ Y, float* __restrict__ Z,
                        int M, int K) {
  int ch = blockIdx.x, mt = blockIdx.y;
  const float* Yc = Y + (size_t)ch * K * 64;
  float* Zc = Z + (size_t)ch * M * 64;
  __shared__ _Float16 sY[32 * 68];
  __shared__ _Float16 sTre[16 * 34]; // TDM: 32 f16 row + 1 dword pad -> stride 34
  __shared__ _Float16 sTim[16 * 34];
  int tid = threadIdx.x, wave = tid >> 5;
  int part = wave >> 1, nt = wave & 1;
  int m0 = mt * 16;
  v8f acc = {};
  for (int k0 = 0; k0 < K; k0 += 32) {
    __syncthreads();
#pragma unroll
    for (int j = 0; j < 4; ++j) {   // stage Y: 32x64 f32 -> f16 (512 f32x4 slots)
      int s = tid + j * 128;
      int r = s >> 4, c4 = (s & 15) * 4;
      f32x4 v = *(const f32x4*)(Yc + (size_t)(k0 + r) * 64 + c4);
      v4h h; h[0] = (_Float16)v[0]; h[1] = (_Float16)v[1];
             h[2] = (_Float16)v[2]; h[3] = (_Float16)v[3];
      *(v4h*)(sY + r * 68 + c4) = h;
    }
    if (wave == 0) {   // twiddle tiles via TDM: pad 1 dword per 16-dword row
      tdm_load_2d_f16(Tre + (size_t)m0 * K + k0, sTre, 32, 16, K, M, 3, 0, true);
      tdm_load_2d_f16(Tim + (size_t)m0 * K + k0, sTim, 32, 16, K, M, 3, 0, true);
      __builtin_amdgcn_s_wait_tensorcnt(0);
    }
    __syncthreads();
    v16h aRe = ldsA(sTre, 34);
    v16h aIm = ldsA(sTim, 34);
    if (part == 0) {  // Zre = Tre*Yre - Tim*Yim (sign folded into B load)
      acc = wmma_f16(aRe, ldsB(sY, 68, nt * 16, false), acc);
      acc = wmma_f16(aIm, ldsB(sY, 68, 32 + nt * 16, true), acc);
    } else {          // Zim = Tre*Yim + Tim*Yre
      acc = wmma_f16(aRe, ldsB(sY, 68, 32 + nt * 16, false), acc);
      acc = wmma_f16(aIm, ldsB(sY, 68, nt * 16, false), acc);
    }
  }
  stC(Zc, 64, m0, part * 32 + nt * 16, acc);
}

// ------------------------- per-mode complex channel mix ---------------------
__global__ void k_mode_mix(const float* __restrict__ Zin, const float* __restrict__ W,
                           float* __restrict__ Zout, int Cin, int Cout) {
  int bt = blockIdx.x, o = blockIdx.y;
  for (int i = threadIdx.x; i < 2048; i += 256) {
    int r = i >> 5, ky = i & 31;
    int blk = r >> 5, m = r & 31;
    float ar = 0.f, ai = 0.f;
    for (int ci = 0; ci < Cin; ++ci) {
      const float* zp = Zin + (((size_t)(bt * Cin + ci)) * 64 + r) * 64;
      float zr = zp[ky], zi = zp[32 + ky];
      const float* wp = W + ((((size_t)((blk * Cout + o) * Cin + ci)) * 32 + m) * 32 + ky) * 2;
      ar += zr * wp[0] - zi * wp[1];
      ai += zr * wp[1] + zi * wp[0];
    }
    float* op = Zout + (((size_t)(bt * Cout + o)) * 64 + r) * 64;
    op[ky] = ar;
    op[32 + ky] = ai;
  }
}

// ------------------------------ instance norm -------------------------------
__global__ void k_inorm(const float* __restrict__ X, float* __restrict__ Y) {
  int ch = blockIdx.x;
  const float* x = X + (size_t)ch * HW;
  float* y = Y + (size_t)ch * HW;
  __shared__ float s1[256], s2[256];
  int tid = threadIdx.x;
  float a = 0.f, b = 0.f;
  for (int i = tid; i < HW; i += 256) { float v = x[i]; a += v; b += v * v; }
  s1[tid] = a; s2[tid] = b; __syncthreads();
  for (int s = 128; s > 0; s >>= 1) {
    if (tid < s) { s1[tid] += s1[tid + s]; s2[tid] += s2[tid + s]; }
    __syncthreads();
  }
  float mean = s1[0] * (1.f / HW);
  float var = s2[0] * (1.f / HW) - mean * mean;
  float rstd = rsqrtf(var + 1e-5f);
  for (int i = tid; i < HW; i += 256) y[i] = (x[i] - mean) * rstd;
}

// -------------------------- attention: Gram via WMMA ------------------------
// One block per (b,h): full padded 48x48 Gram, Q/K read exactly once.
// Wave w owns tiles {w, w+4, w+8} of the 3x3 tile grid (tile = tt*3+st).
__global__ void k_gram(const float* __restrict__ Q, const float* __restrict__ Kb,
                       float* __restrict__ S) {
  int bh = blockIdx.x;
  int b = bh >> 1, h = bh & 1;
  const int D = 4 * HW;
  size_t base = ((size_t)(b * NT) * HC + h * CC) * HW;
  __shared__ _Float16 sQ[48 * 36];
  __shared__ _Float16 sK[48 * 36];
  int tid = threadIdx.x, wave = tid >> 5;
  v8f acc0 = {}, acc1 = {}, acc2 = {};
  for (int k0 = 0; k0 < D; k0 += 32) {
    __syncthreads();
#pragma unroll
    for (int j = 0; j < 3; ++j) {    // stage Q,K chunks: 48x32, f32x4 loads
      int s = tid + j * 128;         // 384 slots
      int r = s >> 3, c4 = (s & 7) * 4;
      v4h hq = {}, hk = {};
      if (r < NT) {
        f32x4 q = *(const f32x4*)(Q + base + (size_t)r * HC * HW + k0 + c4);
        f32x4 k = *(const f32x4*)(Kb + base + (size_t)r * HC * HW + k0 + c4);
        hq[0] = (_Float16)q[0]; hq[1] = (_Float16)q[1];
        hq[2] = (_Float16)q[2]; hq[3] = (_Float16)q[3];
        hk[0] = (_Float16)k[0]; hk[1] = (_Float16)k[1];
        hk[2] = (_Float16)k[2]; hk[3] = (_Float16)k[3];
      }
      *(v4h*)(sQ + r * 36 + c4) = hq;
      *(v4h*)(sK + r * 36 + c4) = hk;
    }
    __syncthreads();
    int t0 = wave;
    acc0 = wmma_f16(ldsA(sQ + (t0 / 3) * 16 * 36, 36), ldsBt(sK, 36, (t0 % 3) * 16), acc0);
    int t1 = wave + 4;
    acc1 = wmma_f16(ldsA(sQ + (t1 / 3) * 16 * 36, 36), ldsBt(sK, 36, (t1 % 3) * 16), acc1);
    if (wave == 0)
      acc2 = wmma_f16(ldsA(sQ + 2 * 16 * 36, 36), ldsBt(sK, 36, 2 * 16), acc2);
  }
  const float scale = 1.0f / 256.0f;  // 1/sqrt(c*H*W)
  float* Sb = S + (size_t)bh * 48 * 48;
  int lane = tid & 31;
  int n = lane & 15, mh = (lane >> 4) * 8;
  {
    int tt = wave / 3, st = wave % 3;
#pragma unroll
    for (int r = 0; r < 8; ++r)
      Sb[(size_t)(tt * 16 + mh + r) * 48 + st * 16 + n] = acc0[r] * scale;
  }
  {
    int t1 = wave + 4, tt = t1 / 3, st = t1 % 3;
#pragma unroll
    for (int r = 0; r < 8; ++r)
      Sb[(size_t)(tt * 16 + mh + r) * 48 + st * 16 + n] = acc1[r] * scale;
  }
  if (wave == 0) {
#pragma unroll
    for (int r = 0; r < 8; ++r)
      Sb[(size_t)(2 * 16 + mh + r) * 48 + 2 * 16 + n] = acc2[r] * scale;
  }
}

__global__ void k_softmax(float* S) {  // grid 16*40, block 64
  int row = blockIdx.x;
  int bh = row / NT, t = row % NT;
  float* Srow = S + (size_t)bh * 48 * 48 + (size_t)t * 48;
  __shared__ float red[64];
  int tid = threadIdx.x;
  float v = (tid < NT) ? Srow[tid] : -3.4e38f;
  red[tid] = v; __syncthreads();
  for (int s = 32; s > 0; s >>= 1) { if (tid < s) red[tid] = fmaxf(red[tid], red[tid + s]); __syncthreads(); }
  float mx = red[0]; __syncthreads();
  float e = (tid < NT) ? __expf(v - mx) : 0.f;
  red[tid] = e; __syncthreads();
  for (int s = 32; s > 0; s >>= 1) { if (tid < s) red[tid] += red[tid + s]; __syncthreads(); }
  float inv = 1.0f / red[0];
  if (tid < 48) Srow[tid] = (tid < NT) ? e * inv : 0.f;
}

// av[b][t][h*c+ci][xy] = sum_s P[b,h,t,s] * v[b,h,s,ci,xy]
__global__ void k_attn_apply(const float* __restrict__ P, const float* __restrict__ V,
                             float* __restrict__ AV) {
  int ch = blockIdx.x;                       // (b*NT+t)*HC + hc
  int xy = blockIdx.y * 256 + threadIdx.x;
  int hc = ch & 7, bt = ch >> 3;
  int b = bt / NT, t = bt % NT;
  int h = hc >> 2;
  const float* Prow = P + ((size_t)(b * 2 + h) * 48 + t) * 48;
  size_t vbase = ((size_t)(b * NT) * HC + hc) * HW + xy;
  float acc = 0.f;
#pragma unroll 8
  for (int s = 0; s < NT; ++s)
    acc += Prow[s] * V[vbase + (size_t)s * HC * HW];
  AV[(size_t)ch * HW + xy] = acc;
}

// ----------------------------- elementwise ops ------------------------------
__global__ void k_resid_gelu(const float* __restrict__ T, const float* __restrict__ Mo,
                             float* __restrict__ Y, int nblk) {
  size_t i = (size_t)blockIdx.x * 256 + threadIdx.x;
  Y[i] = T[i] + gelu_f(Mo[i]);
  (void)nblk;
}

__global__ void k_wscomb(const float* __restrict__ Cc, const float* __restrict__ Ws,
                         const float* __restrict__ Y, float* __restrict__ T) {
  int ch = blockIdx.x;                       // bt*4 + o
  int xy = blockIdx.y * 256 + threadIdx.x;
  int o = ch & 3, bt = ch >> 2;
  float acc = Cc[(size_t)ch * HW + xy];
#pragma unroll
  for (int ci = 0; ci < 4; ++ci)
    acc += Ws[o * 4 + ci] * Y[((size_t)(bt * 4 + ci)) * HW + xy];
  T[(size_t)ch * HW + xy] = gelu_f(acc);
}

// --------------------------- PE spectrum packing ----------------------------
__global__ void k_pe_fill(const float* __restrict__ pe, float* __restrict__ Zpe) {
  int i = blockIdx.x * 256 + threadIdx.x;    // 40*64*64
  if (i >= 163840) return;
  int chan = i >> 12;
  int rc = i & 4095;
  int r = rc >> 6, col = rc & 63;
  int ky = col & 31, im = col >> 5;
  float v = 0.f;
  if (r < 16 && ky < 16)
    v = pe[(((size_t)chan * 16 + r) * 16 + ky) * 2 + im];
  Zpe[i] = v;
}

// -------------------------------- lift / proj -------------------------------
__global__ void k_lift(const float* __restrict__ X, const float* __restrict__ pes,
                       const float* __restrict__ w1, const float* __restrict__ b1,
                       const float* __restrict__ w2, const float* __restrict__ b2,
                       float* __restrict__ T) {
  int bv = blockIdx.x;  // b*5+v
  int b = bv / NV, v = bv % NV;
  int xy = blockIdx.y * 256 + threadIdx.x;
  __shared__ float sw1[64 * 9], sb1[64], sw2[32 * 64], sb2[32];
  int tid = threadIdx.x;
  for (int i = tid; i < 576; i += 256) sw1[i] = w1[i];
  for (int i = tid; i < 64; i += 256) sb1[i] = b1[i];
  for (int i = tid; i < 2048; i += 256) sw2[i] = w2[i];
  for (int i = tid; i < 32; i += 256) sb2[i] = b2[i];
  __syncthreads();
  float in[9];
  in[0] = X[(size_t)bv * HW + xy];
#pragma unroll
  for (int p = 0; p < 8; ++p) in[1 + p] = pes[((size_t)(v * 8 + p)) * HW + xy];
  float h1[64];
#pragma unroll 8
  for (int o = 0; o < 64; ++o) {
    float s = sb1[o];
#pragma unroll
    for (int c = 0; c < 9; ++c) s += sw1[o * 9 + c] * in[c];
    h1[o] = gelu_f(s);
  }
#pragma unroll 4
  for (int j = 0; j < 32; ++j) {
    float s = sb2[j];
#pragma unroll
    for (int o = 0; o < 64; ++o) s += sw2[j * 64 + o] * h1[o];
    int tok = v * 8 + (j >> 2), ci = j & 3;
    T[(((size_t)b * NT + tok) * CC + ci) * HW + xy] = s;
  }
}

__global__ void k_proj(const float* __restrict__ T, const float* __restrict__ w1,
                       const float* __restrict__ b1, const float* __restrict__ w2,
                       const float* __restrict__ b2, float* __restrict__ out) {
  int bv = blockIdx.x; int b = bv / NV, v = bv % NV;
  int xy = blockIdx.y * 256 + threadIdx.x;
  __shared__ float sw1[64 * 32], sb1[64], sw2[64];
  int tid = threadIdx.x;
  for (int i = tid; i < 2048; i += 256) sw1[i] = w1[i];
  for (int i = tid; i < 64; i += 256) { sb1[i] = b1[i]; sw2[i] = w2[i]; }
  __syncthreads();
  float h[32];
#pragma unroll
  for (int j = 0; j < 32; ++j) {
    int tok = v * 8 + (j >> 2), ci = j & 3;
    h[j] = T[(((size_t)b * NT + tok) * CC + ci) * HW + xy];
  }
  float acc = b2[0];
#pragma unroll 4
  for (int o = 0; o < 64; ++o) {
    float s = sb1[o];
#pragma unroll
    for (int j = 0; j < 32; ++j) s += sw1[o * 32 + j] * h[j];
    acc += sw2[o] * gelu_f(s);
  }
  out[(size_t)bv * HW + xy] = acc;
}

// =============================== host driver ================================
extern "C" void kernel_launch(void* const* d_in, const int* in_sizes, int n_in,
                              void* d_out, int out_size, void* d_ws, size_t ws_size,
                              hipStream_t stream) {
  const float* x    = (const float*)d_in[0];
  const float* pe   = (const float*)d_in[1];
  const float* lw1  = (const float*)d_in[2];
  const float* lb1  = (const float*)d_in[3];
  const float* lw2  = (const float*)d_in[4];
  const float* lb2  = (const float*)d_in[5];
  const float* Wq   = (const float*)d_in[6];
  const float* Wk   = (const float*)d_in[7];
  const float* Wv   = (const float*)d_in[8];
  const float* Wm   = (const float*)d_in[9];
  const float* Wc   = (const float*)d_in[10];
  const float* Ws   = (const float*)d_in[11];
  const float* pw1  = (const float*)d_in[12];
  const float* pb1  = (const float*)d_in[13];
  const float* pw2  = (const float*)d_in[14];
  const float* pb2  = (const float*)d_in[15];
  (void)in_sizes; (void)n_in; (void)out_size; (void)ws_size;

  char* wsb = (char*)d_ws;
  size_t off = 0;
  _Float16* tabs = (_Float16*)wsb;
  off = (49152 * sizeof(_Float16) + 255) & ~(size_t)255;
  _Float16* Tf1   = tabs;
  _Float16* Tf2re = tabs + 8192;
  _Float16* Tf2im = tabs + 16384;
  _Float16* Ti1re = tabs + 24576;
  _Float16* Ti1im = tabs + 32768;
  _Float16* Ti2   = tabs + 40960;
  auto F = [&](size_t nf) -> float* {
    float* p = (float*)(wsb + off);
    off = (off + nf * sizeof(float) + 255) & ~(size_t)255;
    return p;
  };
  float* Tbuf = F(20971520);   // t: [B*T*c][HW]
  float* TN   = F(20971520);   // normalized
  float* Yb   = F(20971520);   // y (post attention residual)
  float* Ybig = F(20971520);   // [<=2560][128][64] DFT intermediates
  float* Zt   = F(10485760);   // [<=2560][64][64] fwd spectra
  float* Zm   = F(10485760);   // mixed spectra
  float* Qs   = F(41943040);
  float* Ks   = F(41943040);
  float* Vs   = F(41943040);
  float* AV   = F(41943040);
  float* P1   = F(20971520);   // spatial conv outputs (wm / wc)
  float* pes  = F(655360);     // [nv*pd][HW]
  float* S    = F(36864);      // [16][48][48]

  k_fill_tables<<<192, 256, 0, stream>>>(tabs);

  // ---- positional encoding: pack spectrum -> inverse transform ----
  k_pe_fill<<<640, 256, 0, stream>>>(pe, Zt);
  k_cgemm<<<dim3(40, 8), 128, 0, stream>>>(Ti1re, Ti1im, Zt, Ybig, 128, 64);
  k_gemm_rr<<<dim3(40, 8, 2), 128, 0, stream>>>(Ybig, Ti2, pes, 128, 64, 128);

  // ---- lifting MLP -> token functions ----
  k_lift<<<dim3(40, 64), 256, 0, stream>>>(x, pes, lw1, lb1, lw2, lb2, Tbuf);

  const size_t wqk_s = 131072, wm_s = 131072, wc_s = 65536, ws_s = 16;
  for (int l = 0; l < 4; ++l) {
    // -- norm + forward truncated rfft2 of tn (shared by q,k,v) --
    k_inorm<<<1280, 256, 0, stream>>>(Tbuf, TN);
    k_gemm_rr<<<dim3(1280, 8, 1), 128, 0, stream>>>(TN, Tf1, Ybig, 128, 128, 64);
    k_cgemm<<<dim3(1280, 4), 128, 0, stream>>>(Tf2re, Tf2im, Ybig, Zt, 64, 128);
    // -- q --
    k_mode_mix<<<dim3(320, 8), 256, 0, stream>>>(Zt, Wq + (size_t)l * wqk_s, Zm, 4, 8);
    k_cgemm<<<dim3(2560, 8), 128, 0, stream>>>(Ti1re, Ti1im, Zm, Ybig, 128, 64);
    k_gemm_rr<<<dim3(2560, 8, 2), 128, 0, stream>>>(Ybig, Ti2, Qs, 128, 64, 128);
    // -- k --
    k_mode_mix<<<dim3(320, 8), 256, 0, stream>>>(Zt, Wk + (size_t)l * wqk_s, Zm, 4, 8);
    k_cgemm<<<dim3(2560, 8), 128, 0, stream>>>(Ti1re, Ti1im, Zm, Ybig, 128, 64);
    k_gemm_rr<<<dim3(2560, 8, 2), 128, 0, stream>>>(Ybig, Ti2, Ks, 128, 64, 128);
    // -- v --
    k_mode_mix<<<dim3(320, 8), 256, 0, stream>>>(Zt, Wv + (size_t)l * wqk_s, Zm, 4, 8);
    k_cgemm<<<dim3(2560, 8), 128, 0, stream>>>(Ti1re, Ti1im, Zm, Ybig, 128, 64);
    k_gemm_rr<<<dim3(2560, 8, 2), 128, 0, stream>>>(Ybig, Ti2, Vs, 128, 64, 128);
    // -- attention --
    k_gram<<<16, 128, 0, stream>>>(Qs, Ks, S);
    k_softmax<<<640, 64, 0, stream>>>(S);
    k_attn_apply<<<dim3(2560, 64), 256, 0, stream>>>(S, Vs, AV);
    // -- wm spectral conv on av --
    k_gemm_rr<<<dim3(2560, 8, 1), 128, 0, stream>>>(AV, Tf1, Ybig, 128, 128, 64);
    k_cgemm<<<dim3(2560, 4), 128, 0, stream>>>(Tf2re, Tf2im, Ybig, Zt, 64, 128);
    k_mode_mix<<<dim3(320, 4), 256, 0, stream>>>(Zt, Wm + (size_t)l * wm_s, Zm, 8, 4);
    k_cgemm<<<dim3(1280, 8), 128, 0, stream>>>(Ti1re, Ti1im, Zm, Ybig, 128, 64);
    k_gemm_rr<<<dim3(1280, 8, 2), 128, 0, stream>>>(Ybig, Ti2, P1, 128, 64, 128);
    // y = t + gelu(conv)
    k_resid_gelu<<<81920, 256, 0, stream>>>(Tbuf, P1, Yb, 0);
    // -- wc spectral conv on norm(y) --
    k_inorm<<<1280, 256, 0, stream>>>(Yb, TN);
    k_gemm_rr<<<dim3(1280, 8, 1), 128, 0, stream>>>(TN, Tf1, Ybig, 128, 128, 64);
    k_cgemm<<<dim3(1280, 4), 128, 0, stream>>>(Tf2re, Tf2im, Ybig, Zt, 64, 128);
    k_mode_mix<<<dim3(320, 4), 256, 0, stream>>>(Zt, Wc + (size_t)l * wc_s, Zm, 4, 4);
    k_cgemm<<<dim3(1280, 8), 128, 0, stream>>>(Ti1re, Ti1im, Zm, Ybig, 128, 64);
    k_gemm_rr<<<dim3(1280, 8, 2), 128, 0, stream>>>(Ybig, Ti2, P1, 128, 64, 128);
    // t = gelu(conv + Ws*y)
    k_wscomb<<<dim3(1280, 64), 256, 0, stream>>>(P1, Ws + (size_t)l * ws_s, Yb, Tbuf);
  }

  // ---- projection MLP ----
  k_proj<<<dim3(40, 64), 256, 0, stream>>>(Tbuf, pw1, pb1, pw2, pb2, (float*)d_out);
}
